// invariantMLP_WITH_EMBEDDING_PER_CLASS_ACTOR_80900003987892
// MI455X (gfx1250) — matrix operand order repaired and checked by reading, not compiled, not run on hardware
//
#include <hip/hip_runtime.h>
#include <hip/hip_fp16.h>

typedef __attribute__((ext_vector_type(16))) _Float16 v16h;
typedef __attribute__((ext_vector_type(8)))  _Float16 v8h;
typedef __attribute__((ext_vector_type(8)))  float    v8f;
typedef __attribute__((ext_vector_type(4)))  float    v4f;

#define NBATCH 64
#define NPB    2048
#define MAISLE 32
#define ROWS   (NBATCH*NPB)
#define NSEG   (NBATCH*MAISLE)
#define SLOPE  0.01f

// packed-weight offsets, in halfs (each tile = 32x16 f16 = 512 halfs)
#define PK_LIN1   0        // 16(->32)x128 : 8 tiles
#define PK_LIN2   4096     // 128x128      : 32 tiles
#define PK_LIN3   20480    // 128x64       : 16 tiles
#define PK_STAGE1 28672    // halfs per branch used by stage1 (lin1+lin2+lin3)
#define PK_AE1    28672    // 128x128      : 32 tiles
#define PK_AE2    45056    // 128x64       : 16 tiles
#define PK_STAGE2 24576    // halfs per branch used by stage2 (ae1+ae2)
#define PK_BRANCH 53248
#define PK_HEAD1  106496   // 128x64       : 16 tiles
#define PK_TOTAL  114688

// ---------------------------------------------------------------------------
// Weight packing: dst layout per (ntile,ktile) tile: lane l's 16 halfs are
// W[n = ntile*16 + l%16][k = ktile*32 + (i<8 ? 8*(l/16)+i : 16+8*(l/16)+i-8)], i = 0..15.
// ---------------------------------------------------------------------------
__global__ __launch_bounds__(256) void pack_w(const float* __restrict__ W,
                                              _Float16* __restrict__ dst,
                                              int din, int dout, int KT, int total) {
  int tid = blockIdx.x * blockDim.x + threadIdx.x;
  if (tid >= total) return;
  int tile   = tid >> 9;
  int within = tid & 511;
  int lane = within >> 4, i = within & 15;
  int ntile = tile / KT, ktile = tile % KT;
  int n = ntile * 16 + (lane & 15);
  int k = ktile * 32 + ((i < 8) ? 8 * (lane >> 4) + i : 16 + 8 * (lane >> 4) + (i - 8));
  float v = (k < din) ? W[(size_t)n * din + k] : 0.f;
  dst[tid] = (_Float16)v;
}

__global__ __launch_bounds__(256) void zero_f32(float* __restrict__ p, int n) {
  int t = blockIdx.x * blockDim.x + threadIdx.x;
  if (t < n) p[t] = 0.f;
}

// ---------------------------------------------------------------------------
// Async Global->LDS DMA staging (CDNA5): each instruction moves 16B per lane
// (512B per wave). Tracked by ASYNCcnt; publish with s_wait_asynccnt + barrier.
// Low 32 bits of a flat shared-memory address are the LDS byte offset.
// ---------------------------------------------------------------------------
__device__ __forceinline__ void async_stage(const _Float16* __restrict__ g,
                                            const _Float16* l, int bytes,
                                            int tid, int nthreads) {
  unsigned lbase = (unsigned)(unsigned long long)(const void*)l;
  unsigned long long gbase = (unsigned long long)(const void*)g;
#pragma unroll 4
  for (int o = tid * 16; o < bytes; o += nthreads * 16) {
    unsigned laddr = lbase + (unsigned)o;
    unsigned long long gaddr = gbase + (unsigned long long)o;
    asm volatile("global_load_async_to_lds_b128 %0, %1, off"
                 :: "v"(laddr), "v"(gaddr) : "memory");
  }
}

__device__ __forceinline__ void async_wait_all() {
  asm volatile("s_wait_asynccnt 0x0" ::: "memory");
}

// Load a 16x32 f16 A-tile from LDS (row stride = 128 halfs): per lane the
// layout is two contiguous 8-half runs -> two ds_load_b128.
__device__ __forceinline__ v16h load_A(const _Float16* in, int lane, int k0) {
  const int row = lane & 15, g = lane >> 4;
  const _Float16* p = in + row * 128 + k0 + 8 * g;
  v8h lo = *(const v8h*)p;
  v8h hi = *(const v8h*)(p + 16);
  return __builtin_shufflevector(lo, hi, 0,1,2,3,4,5,6,7,8,9,10,11,12,13,14,15);
}

// One dense layer: 16 rows, din = KT*32, dout = NT*16, via WMMA f32_16x16x32_f16.
// Weights come from LDS (staged by async DMA).
template <int KT, int NT, bool RELU>
__device__ __forceinline__ void mlp_layer(const _Float16* __restrict__ in,
                                          _Float16* __restrict__ out, int outColOff,
                                          const _Float16* __restrict__ wpk,
                                          const float* __restrict__ bias, int lane) {
  const int nIdx = lane & 15, g = lane >> 4;
#pragma unroll
  for (int nt = 0; nt < NT; ++nt) {
    v8f acc = {};
#pragma unroll
    for (int kt = 0; kt < KT; ++kt) {
      v16h a = load_A(in, lane, kt * 32);
      v16h b = *(const v16h*)(wpk + (size_t)(nt * KT + kt) * 512 + lane * 16);
      acc = __builtin_amdgcn_wmma_f32_16x16x32_f16(false, a, false, b, (short)0,
                                                   acc, false, false);
    }
    float bv = bias[nt * 16 + nIdx];
#pragma unroll
    for (int r = 0; r < 8; ++r) {
      float v = acc[r] + bv;
      if (RELU) v = fmaxf(v, SLOPE * v);  // lrelu, slope > 0
      out[(r + 8 * g) * 128 + outColOff + nt * 16 + nIdx] = (_Float16)v;
    }
  }
}

// ---------------------------------------------------------------------------
// Stage 1: lin1->lin2->lin3 for both branches; write e (f16), accumulate
// segment sums (LDS ds_add_f32 -> global_atomic_add_f32) and counts.
// Block = 128 threads = 4 waves = 64 rows (all within one batch).
// ---------------------------------------------------------------------------
__global__ __launch_bounds__(128) void stage1_kernel(
    const float* __restrict__ x, const int* __restrict__ aisle,
    const _Float16* __restrict__ wpack, _Float16* __restrict__ ebuf,
    float* __restrict__ gsum, float* __restrict__ gcnt,
    const float* __restrict__ bp1, const float* __restrict__ bp2,
    const float* __restrict__ bp3, const float* __restrict__ bf1,
    const float* __restrict__ bf2, const float* __restrict__ bf3) {
  __shared__ __align__(16) _Float16 actA[4][16 * 128];
  __shared__ __align__(16) _Float16 actB[4][16 * 128];
  __shared__ __align__(16) _Float16 wbuf[PK_STAGE1];  // 57344 B, one branch
  __shared__ float ssum[2][MAISLE][64];
  __shared__ float scnt[MAISLE];

  const int tid = threadIdx.x;
  const int lane = tid & 31;
  const int w = tid >> 5;
  const int row = lane & 15;
  const int g = lane >> 4;
  const int chunk = g;

  for (int i = tid; i < 2 * MAISLE * 64; i += 128) ((float*)ssum)[i] = 0.f;
  if (tid < MAISLE) scnt[tid] = 0.f;
  __syncthreads();
  if (tid < 64) atomicAdd(&scnt[aisle[blockIdx.x * 64 + tid]], 1.f);

  const int rowbase = blockIdx.x * 64 + w * 16;
  const int grow = rowbase + row;

#pragma unroll
  for (int br = 0; br < 2; ++br) {
    if (br) __syncthreads();  // protect wbuf from overwrite while still read
    async_stage(wpack + (size_t)br * PK_BRANCH, wbuf, PK_STAGE1 * 2, tid, 128);
    async_wait_all();
    __syncthreads();

    const float* bb1 = br ? bf1 : bp1;
    const float* bb2 = br ? bf2 : bp2;
    const float* bb3 = br ? bf3 : bp3;
    const int coff = br * 16;

    // L1: A tile built directly from global x (K 0..15 real, 16..31 zero-pad)
    const float* xr = x + (size_t)grow * 32 + coff + 8 * g;
    v4f xa = *(const v4f*)xr;
    v4f xb = *(const v4f*)(xr + 4);
    v16h a1 = {};
#pragma unroll
    for (int q = 0; q < 4; ++q) {
      a1[q]     = (_Float16)xa[q];
      a1[4 + q] = (_Float16)xb[q];
    }
#pragma unroll
    for (int nt = 0; nt < 8; ++nt) {
      v8f acc = {};
      v16h bm = *(const v16h*)(wbuf + PK_LIN1 + (size_t)nt * 512 + lane * 16);
      acc = __builtin_amdgcn_wmma_f32_16x16x32_f16(false, a1, false, bm, (short)0,
                                                   acc, false, false);
      float bv = bb1[nt * 16 + row];
#pragma unroll
      for (int r = 0; r < 8; ++r) {
        float v = acc[r] + bv;
        v = fmaxf(v, SLOPE * v);
        actA[w][(r + 8 * g) * 128 + nt * 16 + row] = (_Float16)v;
      }
    }
    mlp_layer<4, 8, true >(actA[w], actB[w], 0, wbuf + PK_LIN2, bb2, lane);  // lin2
    mlp_layer<4, 4, false>(actB[w], actA[w], 0, wbuf + PK_LIN3, bb3, lane);  // lin3

    // store e tile (64 halfs/row, 2 lanes per row) + LDS segment accumulate
    const _Float16* esrc = &actA[w][row * 128 + chunk * 32];
    _Float16* edst = ebuf + ((size_t)br * ROWS + grow) * 64 + chunk * 32;
#pragma unroll
    for (int q = 0; q < 4; ++q) *(v8h*)(edst + 8 * q) = *(const v8h*)(esrc + 8 * q);

    const int aid = aisle[grow];
    float* sdst = &ssum[br][aid][chunk * 32];
#pragma unroll
    for (int i = 0; i < 32; ++i) atomicAdd(&sdst[i], (float)esrc[i]);
  }

  __syncthreads();
  const int batch = blockIdx.x >> 5;  // 32 blocks (64 rows each) per batch
  for (int i = tid; i < 2 * MAISLE * 64; i += 128) {
    int brr = i >> 11;
    int rem = i & 2047;
    int a = rem >> 6, col = rem & 63;
    atomicAdd(&gsum[((size_t)brr * NSEG + batch * MAISLE + a) * 64 + col],
              ((const float*)ssum)[i]);
  }
  if (tid < MAISLE) atomicAdd(&gcnt[batch * MAISLE + tid], scnt[tid]);
}

__global__ __launch_bounds__(256) void mean_kernel(const float* __restrict__ gsum,
                                                   const float* __restrict__ gcnt,
                                                   _Float16* __restrict__ meanb,
                                                   int total) {
  int i = blockIdx.x * blockDim.x + threadIdx.x;
  if (i >= total) return;
  int seg = (i >> 6) & (NSEG - 1);
  meanb[i] = (_Float16)(gsum[i] / fmaxf(gcnt[seg], 1.f));
}

// ---------------------------------------------------------------------------
// Stage 2: concat[e, mean] -> ae1 -> ae2 (x2 branches) -> concat -> head1 ->
// head2 (per-row dot). Block = 128 threads = 4 waves = 64 rows.
// ---------------------------------------------------------------------------
__global__ __launch_bounds__(128) void stage2_kernel(
    const _Float16* __restrict__ ebuf, const _Float16* __restrict__ meanb,
    const int* __restrict__ aisle, const _Float16* __restrict__ wpack,
    float* __restrict__ logits,
    const float* __restrict__ bpa1, const float* __restrict__ bpa2,
    const float* __restrict__ bfa1, const float* __restrict__ bfa2,
    const float* __restrict__ bh1, const float* __restrict__ wh2,
    const float* __restrict__ bh2) {
  __shared__ __align__(16) _Float16 actA[4][16 * 128];
  __shared__ __align__(16) _Float16 actB[4][16 * 128];
  __shared__ __align__(16) _Float16 hcat[4][16 * 128];
  __shared__ __align__(16) _Float16 wbuf[PK_STAGE2];  // 49152 B, one branch
  __shared__ __align__(16) _Float16 hbuf[8192];       // 16384 B, head1

  const int tid = threadIdx.x, lane = tid & 31, w = tid >> 5;
  const int row = lane & 15, g = lane >> 4, chunk = g;
  const int rowbase = blockIdx.x * 64 + w * 16;
  const int grow = rowbase + row;
  const int seg = (grow >> 11) * MAISLE + aisle[grow];

  // head1 weights staged once; first branch's wait covers this too
  async_stage(wpack + PK_HEAD1, hbuf, 8192 * 2, tid, 128);

#pragma unroll
  for (int br = 0; br < 2; ++br) {
    if (br) __syncthreads();  // protect wbuf from overwrite while still read
    async_stage(wpack + (size_t)br * PK_BRANCH + PK_AE1, wbuf, PK_STAGE2 * 2,
                tid, 128);
    async_wait_all();
    __syncthreads();

    const float* ba1 = br ? bfa1 : bpa1;
    const float* ba2 = br ? bfa2 : bpa2;

    const _Float16* esrc = ebuf + ((size_t)br * ROWS + grow) * 64 + chunk * 32;
    const _Float16* msrc = meanb + ((size_t)br * NSEG + seg) * 64 + chunk * 32;
    _Float16* d0 = &actA[w][row * 128 + chunk * 32];
    _Float16* d1 = &actA[w][row * 128 + 64 + chunk * 32];
#pragma unroll
    for (int q = 0; q < 4; ++q) {
      *(v8h*)(d0 + 8 * q) = *(const v8h*)(esrc + 8 * q);
      *(v8h*)(d1 + 8 * q) = *(const v8h*)(msrc + 8 * q);
    }
    mlp_layer<4, 8, true>(actA[w], actB[w], 0, wbuf, ba1, lane);              // ae1
    mlp_layer<4, 4, true>(actB[w], hcat[w], br * 64,
                          wbuf + (PK_AE2 - PK_AE1), ba2, lane);               // ae2
  }
  mlp_layer<4, 4, true>(hcat[w], actB[w], 0, hbuf, bh1, lane);                // head1

  // head2: dot(64) per row; two lanes per row, combine via shfl_xor(16)
  float sum = 0.f;
#pragma unroll
  for (int i = 0; i < 32; ++i)
    sum += (float)actB[w][row * 128 + chunk * 32 + i] * wh2[chunk * 32 + i];
  sum += __shfl_xor(sum, 16);
  if (lane < 16) logits[rowbase + lane] = sum + bh2[0];
}

__global__ __launch_bounds__(256) void softmax_kernel(const float* __restrict__ logits,
                                                      const int* __restrict__ mask,
                                                      float* __restrict__ out) {
  __shared__ float red[256];
  const int b = blockIdx.x, t = threadIdx.x;
  const float* lrow = logits + (size_t)b * NPB;
  const int* mrow = mask + (size_t)b * NPB;
  float* orow = out + (size_t)b * NPB;

  float mx = -__builtin_inff();
  for (int i = t; i < NPB; i += 256) {
    float v = (mrow[i] != 0) ? lrow[i] : -__builtin_inff();
    mx = fmaxf(mx, v);
  }
  red[t] = mx;
  __syncthreads();
  for (int s = 128; s > 0; s >>= 1) {
    if (t < s) red[t] = fmaxf(red[t], red[t + s]);
    __syncthreads();
  }
  mx = red[0];
  __syncthreads();

  float sum = 0.f;
  for (int i = t; i < NPB; i += 256) {
    float v = (mrow[i] != 0) ? __expf(lrow[i] - mx) : 0.f;
    orow[i] = v;
    sum += v;
  }
  red[t] = sum;
  __syncthreads();
  for (int s = 128; s > 0; s >>= 1) {
    if (t < s) red[t] += red[t + s];
    __syncthreads();
  }
  const float inv = 1.f / red[0];
  for (int i = t; i < NPB; i += 256) orow[i] *= inv;
}

// ---------------------------------------------------------------------------
extern "C" void kernel_launch(void* const* d_in, const int* in_sizes, int n_in,
                              void* d_out, int out_size, void* d_ws, size_t ws_size,
                              hipStream_t stream) {
  const float* x     = (const float*)d_in[0];
  const int*   aisle = (const int*)d_in[1];
  // d_in[2] = batch_ids (implied by row index; unused)
  const int*   mask  = (const int*)d_in[3];
  // params in dict order: perf lin1/lin2/lin3/ae1/ae2 (W,b), fair same, head1, head2
  const float* pW[12];
  for (int i = 0; i < 12; ++i) pW[i] = (const float*)d_in[4 + 2 * i];  // weights
  const float* pB[12];
  for (int i = 0; i < 12; ++i) pB[i] = (const float*)d_in[5 + 2 * i];  // biases
  // indices: 0..4 perf lin1,lin2,lin3,ae1,ae2 ; 5..9 fair ; 10 head1 ; 11 head2

  char* ws = (char*)d_ws;
  size_t off = 0;
  auto take = [&](size_t bytes) -> char* {
    char* p = ws + off;
    off += (bytes + 255) & ~(size_t)255;
    return p;
  };
  _Float16* wpack = (_Float16*)take((size_t)PK_TOTAL * 2);
  _Float16* ebuf  = (_Float16*)take((size_t)2 * ROWS * 64 * 2);
  float* gsum     = (float*)take((size_t)2 * NSEG * 64 * 4);  // 256B-multiple,
  float* gcnt     = (float*)take((size_t)NSEG * 4);           // so contiguous
  _Float16* meanb = (_Float16*)take((size_t)2 * NSEG * 64 * 2);
  float* logits   = (float*)take((size_t)ROWS * 4);

  auto packL = [&](const float* W, size_t dstHalfs, int din, int dout, int KT) {
    int total = (dout / 16) * KT * 512;
    pack_w<<<(total + 255) / 256, 256, 0, stream>>>(W, wpack + dstHalfs, din, dout,
                                                    KT, total);
  };
  for (int br = 0; br < 2; ++br) {
    size_t base = (size_t)br * PK_BRANCH;
    packL(pW[br * 5 + 0], base + PK_LIN1, 16, 128, 1);
    packL(pW[br * 5 + 1], base + PK_LIN2, 128, 128, 4);
    packL(pW[br * 5 + 2], base + PK_LIN3, 128, 64, 4);
    packL(pW[br * 5 + 3], base + PK_AE1, 128, 128, 4);
    packL(pW[br * 5 + 4], base + PK_AE2, 128, 64, 4);
  }
  packL(pW[10], PK_HEAD1, 128, 64, 4);

  const int nzero = 2 * NSEG * 64 + NSEG;  // gsum + gcnt (contiguous)
  zero_f32<<<(nzero + 255) / 256, 256, 0, stream>>>(gsum, nzero);

  stage1_kernel<<<ROWS / 64, 128, 0, stream>>>(x, aisle, wpack, ebuf, gsum, gcnt,
                                               pB[0], pB[1], pB[2],
                                               pB[5], pB[6], pB[7]);

  const int nmean = 2 * NSEG * 64;
  mean_kernel<<<(nmean + 255) / 256, 256, 0, stream>>>(gsum, gcnt, meanb, nmean);

  stage2_kernel<<<ROWS / 64, 128, 0, stream>>>(ebuf, meanb, aisle, wpack, logits,
                                               pB[3], pB[4], pB[8], pB[9],
                                               pB[10], pW[11], pB[11]);

  softmax_kernel<<<NBATCH, 256, 0, stream>>>(logits, mask, (float*)d_out);
}
